// QwenAttention_32083405701330
// MI455X (gfx1250) — matrix operand order repaired and checked by reading, not compiled, not run on hardware
//
#include <hip/hip_runtime.h>
#include <hip/hip_bf16.h>
#include <math.h>
#include <stdint.h>

// ---------------- problem constants ----------------
#define HIDDEN  2048
#define NHEADS  16
#define NKV     2
#define HDIM    128
#define BB      2
#define LL      2048
#define MM      (BB * LL)        // 4096 rows of x
#define EKV     (NKV * HDIM)     // 256

typedef __bf16 bf16_t;
typedef __attribute__((ext_vector_type(16))) __bf16 v16bf;
typedef __attribute__((ext_vector_type(8)))  __bf16 v8bf;
typedef __attribute__((ext_vector_type(8)))  float  v8f;

// ---------------- WMMA wrapper ----------------
__device__ __forceinline__ v8f wmma_bf16(v16bf a, v16bf b, v8f c) {
  // D = A(16x32 bf16) * B(32x16 bf16) + C(16x16 f32)
  return __builtin_amdgcn_wmma_f32_16x16x32_bf16(
      /*neg_a=*/false, a, /*neg_b=*/false, b,
      /*c_mod=*/(short)0, c, /*reuse_a=*/false, /*reuse_b=*/false);
}

// A-fragment (16x32, row-major source, K-contiguous).
// Lane m (0..15): elems[0..7]=A[m][kb..kb+7], elems[8..15]=A[m][kb+16..kb+23]
// Lane m+16:      elems[0..7]=A[m][kb+8..+15], elems[8..15]=A[m][kb+24..+31]
__device__ __forceinline__ v16bf ld_frag_a(const bf16_t* base, int ld,
                                           int mbase, int kb, int lane) {
  int row  = mbase + (lane & 15);
  int koff = kb + ((lane >> 4) ? 8 : 0);
  const bf16_t* p = base + (size_t)row * ld + koff;
  union { v16bf v; v8bf h[2]; } u;
  u.h[0] = *(const v8bf*)p;         // K = koff .. koff+7
  u.h[1] = *(const v8bf*)(p + 16);  // K = koff+16 .. koff+23
  return u.v;
}

// B-fragment (32x16). B[k][n] taken from row n of a [N,K] row-major matrix.
// Lane n (0..15): elems[0..7]=K kb..kb+7, elems[8..15]=K kb+16..kb+23; lane n+16: kb+8..+15, kb+24..+31.
__device__ __forceinline__ v16bf ld_frag_b(const bf16_t* base, int ld,
                                           int nbase, int kb, int lane) {
  int row  = nbase + (lane & 15);
  int koff = kb + ((lane >> 4) ? 8 : 0);
  const bf16_t* p = base + (size_t)row * ld + koff;
  union { v16bf v; v8bf h[2]; } u;
  u.h[0] = *(const v8bf*)p;
  u.h[1] = *(const v8bf*)(p + 16);
  return u.v;
}

// ---------------- async global -> LDS copy (gfx1250 ASYNCcnt path) ---------
__device__ __forceinline__ void async_b128(uint32_t lds_addr,
                                           const bf16_t* gbase, uint32_t voff) {
  // LDS address from the VDST operand; global address = SGPR base + 32-bit VGPR byte offset
  asm volatile("global_load_async_to_lds_b128 %0, %1, %2"
               :: "v"(lds_addr), "v"(voff),
                  "s"((unsigned long long)(uintptr_t)gbase)
               : "memory");
}

// Stage a 128x32 A tile and a 64x32 B tile (bf16) into LDS.
// 256 threads: 2 async b128 per thread for A, 1 for B (3 ASYNCcnt per wave).
__device__ __forceinline__ void stage_tiles(const bf16_t* A, const bf16_t* W,
                                            int K, int mblock, int nblock, int kb,
                                            bf16_t (*sAb)[32], bf16_t (*sBb)[32],
                                            int tid) {
  const bf16_t* ga = A + (size_t)mblock * K + kb;
  const bf16_t* gb = W + (size_t)nblock * K + kb;
  uint32_t ldsA = (uint32_t)(uintptr_t)&sAb[0][0];
  uint32_t ldsB = (uint32_t)(uintptr_t)&sBb[0][0];
#pragma unroll
  for (int j = 0; j < 2; ++j) {
    int q = tid * 2 + j;                  // 0..511 chunks of 16B
    int row = q >> 2, c16 = q & 3;
    async_b128(ldsA + (uint32_t)(row * 32 + c16 * 8) * 2, ga,
               (uint32_t)(row * K + c16 * 8) * 2);
  }
  {
    int row = tid >> 2, c16 = tid & 3;    // 256 chunks of 16B
    async_b128(ldsB + (uint32_t)(row * 32 + c16 * 8) * 2, gb,
               (uint32_t)(row * K + c16 * 8) * 2);
  }
}

// ---------------- fp32 -> bf16 conversion ----------------
__global__ void cvt_f32_bf16(const float* __restrict__ in,
                             bf16_t* __restrict__ out, int n) {
  int i = blockIdx.x * blockDim.x + threadIdx.x;
  if (i < n) out[i] = (bf16_t)in[i];
}

// ---------------- V transpose: [B,L,EKV] -> [B,NKV,HDIM,L] ----------------
__global__ void transpose_v(const bf16_t* __restrict__ vr,
                            bf16_t* __restrict__ vt) {
  int idx = blockIdx.x * blockDim.x + threadIdx.x;
  if (idx >= MM * EKV) return;
  int b = idx / (LL * EKV);
  int r = idx % (LL * EKV);
  int l = r / EKV;
  int c = r % EKV;
  int kvh = c / HDIM, d = c % HDIM;
  vt[(((size_t)b * NKV + kvh) * HDIM + d) * LL + l] = vr[idx];
}

// -------- WMMA GEMM, async-LDS double-buffered: C = A * W^T (+bias) --------
// block = 256 threads = 8 waves (4 in M x 2 in N); block tile 128x64, K step 32
template <bool HAS_BIAS, bool OUT_F32>
__global__ __launch_bounds__(256)
void gemm_wmma_async(const bf16_t* __restrict__ A, const bf16_t* __restrict__ W,
                     const float* __restrict__ bias, void* __restrict__ Cout,
                     int N, int K) {
  __shared__ bf16_t sA[2][128][32];     // 16 KB
  __shared__ bf16_t sB[2][64][32];      //  8 KB
  const int tid  = threadIdx.x;
  const int wave = tid >> 5, lane = tid & 31;
  const int wm = wave & 3, wn = wave >> 2;
  const int mblock = blockIdx.x * 128;
  const int nblock = blockIdx.y * 64;

  v8f acc00 = {}, acc01 = {}, acc10 = {}, acc11 = {};
  const int nk = K / 32;

  stage_tiles(A, W, K, mblock, nblock, 0, sA[0], sB[0], tid);

  for (int i = 0; i < nk; ++i) {
    const int cur = i & 1;
    if (i + 1 < nk) {
      stage_tiles(A, W, K, mblock, nblock, (i + 1) * 32,
                  sA[cur ^ 1], sB[cur ^ 1], tid);
      asm volatile("s_wait_asynccnt 0x3" ::: "memory");  // step i landed
    } else {
      asm volatile("s_wait_asynccnt 0x0" ::: "memory");
    }
    __syncthreads();                       // tile i visible to all waves

    const bf16_t* Ab = &sA[cur][0][0];
    const bf16_t* Bb = &sB[cur][0][0];
    v16bf a0 = ld_frag_a(Ab, 32, wm * 32,      0, lane);
    v16bf a1 = ld_frag_a(Ab, 32, wm * 32 + 16, 0, lane);
    v16bf b0 = ld_frag_b(Bb, 32, wn * 32,      0, lane);
    v16bf b1 = ld_frag_b(Bb, 32, wn * 32 + 16, 0, lane);
    acc00 = wmma_bf16(a0, b0, acc00);
    acc01 = wmma_bf16(a0, b1, acc01);
    acc10 = wmma_bf16(a1, b0, acc10);
    acc11 = wmma_bf16(a1, b1, acc11);

    __syncthreads();                       // done reading buf before overwrite
  }

  const int mbase = mblock + wm * 32;
  const int nbase = nblock + wn * 32;
  const int lgrp = lane >> 4, ln = lane & 15;
  v8f* accs[2][2] = {{&acc00, &acc01}, {&acc10, &acc11}};
#pragma unroll
  for (int tm = 0; tm < 2; ++tm) {
#pragma unroll
    for (int tn = 0; tn < 2; ++tn) {
      int col = nbase + tn * 16 + ln;
      float bv = HAS_BIAS ? bias[col] : 0.0f;
      v8f a = *accs[tm][tn];
#pragma unroll
      for (int r = 0; r < 8; ++r) {
        int row = mbase + tm * 16 + r + lgrp * 8;
        float v = a[r] + bv;
        if (OUT_F32)
          ((float*)Cout)[(size_t)row * N + col] = v;
        else
          ((bf16_t*)Cout)[(size_t)row * N + col] = (bf16_t)v;
      }
    }
  }
}

// ---------------- flash attention (causal, GQA) ----------------
// grid.x = B*NHEADS*(LL/64); block = 128 threads = 4 waves.
// Each wave: one 16-row query tile. K-blocks of 32.
__global__ __launch_bounds__(128)
void attn_flash(const bf16_t* __restrict__ Q,   // [B,L,HIDDEN], head h at col h*128
                const bf16_t* __restrict__ Kc,  // [B,L,EKV]
                const bf16_t* __restrict__ Vt,  // [B,NKV,HDIM,L]
                bf16_t* __restrict__ O) {       // [B,L,HIDDEN]
  const int wave = threadIdx.x >> 5;
  const int lane = threadIdx.x & 31;
  const int lgrp = lane >> 4, ln = lane & 15;

  const int nq64 = LL / 64;
  const int bh = blockIdx.x / nq64;
  const int q64 = blockIdx.x % nq64;
  const int b = bh / NHEADS, h = bh % NHEADS, kvh = h % NKV;
  const int qbase = q64 * 64 + wave * 16;

  const bf16_t* Qp = Q  + (size_t)b * LL * HIDDEN + h * HDIM;
  const bf16_t* Kp = Kc + (size_t)b * LL * EKV + kvh * HDIM;
  const bf16_t* Vp = Vt + (size_t)(b * NKV + kvh) * HDIM * LL;

  // Q fragments for the whole D=128 (4 chunks of 32), kept resident.
  v16bf qf[4];
#pragma unroll
  for (int c = 0; c < 4; ++c) qf[c] = ld_frag_a(Qp, HIDDEN, qbase, c * 32, lane);

  v8f o[8];
#pragma unroll
  for (int t = 0; t < 8; ++t) o[t] = (v8f){};
  float mrow[8], lrow[8];
#pragma unroll
  for (int r = 0; r < 8; ++r) { mrow[r] = -__builtin_inff(); lrow[r] = 0.0f; }

  __shared__ bf16_t pbuf[4][16][32];            // 1 KB per wave
  bf16_t (*P)[32] = pbuf[wave];

  const float scale = 0.08838834764831845f;     // 1/sqrt(128)
  const int nkb = ((qbase + 15) >> 5) + 1;      // causal: cover k <= qbase+15

  for (int kb32 = 0; kb32 < nkb; ++kb32) {
    const int kb = kb32 * 32;

    // prefetch next K/V block rows (global_prefetch_b8)
    if (kb + 32 < LL) {
      __builtin_prefetch(Kp + (size_t)(kb + 32 + lane) * EKV, 0, 0);
      __builtin_prefetch(Vp + (size_t)(lane) * LL + kb + 32, 0, 0);
    }

    // S(16x32) = Q * K^T
    v8f s0 = {}, s1 = {};
#pragma unroll
    for (int c = 0; c < 4; ++c) {
      v16bf kb0 = ld_frag_b(Kp, EKV, kb,      c * 32, lane);
      v16bf kb1 = ld_frag_b(Kp, EKV, kb + 16, c * 32, lane);
      s0 = wmma_bf16(qf[c], kb0, s0);
      s1 = wmma_bf16(qf[c], kb1, s1);
    }

    // scale + causal mask
#pragma unroll
    for (int r = 0; r < 8; ++r) {
      int row = qbase + r + lgrp * 8;
      int c0 = kb + ln, c1 = kb + 16 + ln;
      float v0 = s0[r] * scale, v1 = s1[r] * scale;
      if (c0 > row) v0 = -3.0e38f;
      if (c1 > row) v1 = -3.0e38f;
      s0[r] = v0; s1[r] = v1;
    }

    // online softmax per row (rows live across the 16-lane half-groups)
#pragma unroll
    for (int r = 0; r < 8; ++r) {
      float mx = fmaxf(s0[r], s1[r]);
#pragma unroll
      for (int off = 8; off >= 1; off >>= 1)
        mx = fmaxf(mx, __shfl_xor(mx, off, 16));
      float mnew  = fmaxf(mrow[r], mx);
      float alpha = __expf(mrow[r] - mnew);
      float p0 = __expf(s0[r] - mnew);
      float p1 = __expf(s1[r] - mnew);
      P[r + lgrp * 8][ln]      = (bf16_t)p0;
      P[r + lgrp * 8][16 + ln] = (bf16_t)p1;
      float rs = p0 + p1;
#pragma unroll
      for (int off = 8; off >= 1; off >>= 1)
        rs += __shfl_xor(rs, off, 16);
      lrow[r] = lrow[r] * alpha + rs;
      mrow[r] = mnew;
#pragma unroll
      for (int t = 0; t < 8; ++t) o[t][r] *= alpha;
    }

    // wave-local LDS RAW: split dep counter on gfx1250
    asm volatile("s_wait_dscnt 0" ::: "memory");

    // Re-read P in A-fragment layout (16x32 over k)
    union { v16bf v; v8bf hv[2]; } pu;
    {
      const v8bf* prow = (const v8bf*)&P[ln][0];  // 4 chunks of 8 per row
      pu.hv[0] = prow[lgrp];         // k = 0..7  (grp0) / 8..15  (grp1)
      pu.hv[1] = prow[lgrp + 2];     // k = 16..23 (grp0) / 24..31 (grp1)
    }
    asm volatile("" ::: "memory");

    // O(16x128) += P(16x32) * V(32x128)  — V^T rows are d, contiguous in L
#pragma unroll
    for (int t = 0; t < 8; ++t) {
      v16bf vf = ld_frag_b(Vp, LL, t * 16, kb, lane);
      o[t] = wmma_bf16(pu.v, vf, o[t]);
    }
  }

  // normalize and store (C-layout)
#pragma unroll
  for (int t = 0; t < 8; ++t) {
    int col = h * HDIM + t * 16 + ln;
#pragma unroll
    for (int r = 0; r < 8; ++r) {
      int row = qbase + r + lgrp * 8;
      float v = o[t][r] / lrow[r];
      O[((size_t)b * LL + row) * HIDDEN + col] = (bf16_t)v;
    }
  }
}

// ---------------- host launcher ----------------
extern "C" void kernel_launch(void* const* d_in, const int* in_sizes, int n_in,
                              void* d_out, int out_size, void* d_ws, size_t ws_size,
                              hipStream_t stream) {
  (void)in_sizes; (void)n_in; (void)out_size; (void)ws_size;
  const float* x  = (const float*)d_in[0];
  const float* wq = (const float*)d_in[1];
  const float* bq = (const float*)d_in[2];
  const float* wk = (const float*)d_in[3];
  const float* bk = (const float*)d_in[4];
  const float* wv = (const float*)d_in[5];
  const float* bv = (const float*)d_in[6];
  const float* wo = (const float*)d_in[7];

  char* ws = (char*)d_ws;
  size_t off = 0;
  auto carve = [&](size_t bytes) { char* p = ws + off; off += (bytes + 255) & ~(size_t)255; return p; };
  bf16_t* xb   = (bf16_t*)carve((size_t)MM * HIDDEN * 2);       // 16 MiB
  bf16_t* wqb  = (bf16_t*)carve((size_t)HIDDEN * HIDDEN * 2);   //  8 MiB
  bf16_t* wkb  = (bf16_t*)carve((size_t)EKV * HIDDEN * 2);      //  1 MiB
  bf16_t* wvb  = (bf16_t*)carve((size_t)EKV * HIDDEN * 2);      //  1 MiB
  bf16_t* wob  = (bf16_t*)carve((size_t)HIDDEN * HIDDEN * 2);   //  8 MiB
  bf16_t* Qb   = (bf16_t*)carve((size_t)MM * HIDDEN * 2);       // 16 MiB
  bf16_t* Kb   = (bf16_t*)carve((size_t)MM * EKV * 2);          //  2 MiB
  bf16_t* Vraw = (bf16_t*)carve((size_t)MM * EKV * 2);          //  2 MiB
  bf16_t* Vt   = (bf16_t*)carve((size_t)MM * EKV * 2);          //  2 MiB
  bf16_t* Ob   = (bf16_t*)carve((size_t)MM * HIDDEN * 2);       // 16 MiB

  auto cvt = [&](const float* src, bf16_t* dst, int n) {
    cvt_f32_bf16<<<(n + 255) / 256, 256, 0, stream>>>(src, dst, n);
  };
  cvt(x,  xb,  MM * HIDDEN);
  cvt(wq, wqb, HIDDEN * HIDDEN);
  cvt(wk, wkb, EKV * HIDDEN);
  cvt(wv, wvb, EKV * HIDDEN);
  cvt(wo, wob, HIDDEN * HIDDEN);

  // QKV projections: C = xb * W^T + bias
  {
    dim3 g(MM / 128, HIDDEN / 64);
    gemm_wmma_async<true, false><<<g, 256, 0, stream>>>(xb, wqb, bq, Qb, HIDDEN, HIDDEN);
  }
  {
    dim3 g(MM / 128, EKV / 64);
    gemm_wmma_async<true, false><<<g, 256, 0, stream>>>(xb, wkb, bk, Kb, EKV, HIDDEN);
    gemm_wmma_async<true, false><<<g, 256, 0, stream>>>(xb, wvb, bv, Vraw, EKV, HIDDEN);
  }
  transpose_v<<<(MM * EKV + 255) / 256, 256, 0, stream>>>(Vraw, Vt);

  // attention
  attn_flash<<<BB * NHEADS * (LL / 64), 128, 0, stream>>>(Qb, Kb, Vt, Ob);

  // output projection: d_out(f32) = Ob * wo^T
  {
    dim3 g(MM / 128, HIDDEN / 64);
    gemm_wmma_async<false, true><<<g, 256, 0, stream>>>(Ob, wob, nullptr, (float*)d_out,
                                                        HIDDEN, HIDDEN);
  }
}
